// ResnetBlock_75849122448158
// MI455X (gfx1250) — matrix operand order repaired and checked by reading, not compiled, not run on hardware
//

#include <hip/hip_runtime.h>
#include <hip/hip_bf16.h>
#include <math.h>

typedef __attribute__((ext_vector_type(16))) _Float16 v16h;
typedef __attribute__((ext_vector_type(8)))  _Float16 v8h;
typedef __attribute__((ext_vector_type(8)))  float    v8f;
typedef __attribute__((ext_vector_type(4)))  unsigned int uint32x4;
typedef __attribute__((ext_vector_type(8)))  int      int32x8;
typedef __attribute__((ext_vector_type(4)))  int      int32x4;

#define Bsz 16
#define Cch 128
#define Hh  128
#define Ww  128

// LDS staging geometry: 4 input rows x 130 pixels (halo) x 136-half stride
// (128 data halves + 8 pad halves so per-pixel stride = 272B -> conflict-free
// ds_load_b128 across 16 lanes reading consecutive pixels).
#define PIX_STRIDE 136
#define ROW_PIX    130
#define SMEM_HALVES (4 * ROW_PIX * PIX_STRIDE)   // 70720 halves = 141440 B

// ---------------------------------------------------------------------------
// GroupNorm pass 1: per (b, group) mean / rstd.
// ---------------------------------------------------------------------------
__global__ __launch_bounds__(256)
void gn_stats(const float* __restrict__ x, float* __restrict__ meanr,
              float* __restrict__ rstdr) {
  int bg = blockIdx.x;                      // 0..511  (b*32 + g)
  const float* p = x + (size_t)bg * 65536;
  float s = 0.f, q = 0.f;
  for (int i = threadIdx.x; i < 65536; i += 256) {
    float v = p[i];
    s += v; q += v * v;
  }
  __shared__ float ss[256], qq[256];
  ss[threadIdx.x] = s; qq[threadIdx.x] = q;
  __syncthreads();
  for (int st = 128; st > 0; st >>= 1) {
    if (threadIdx.x < st) {
      ss[threadIdx.x] += ss[threadIdx.x + st];
      qq[threadIdx.x] += qq[threadIdx.x + st];
    }
    __syncthreads();
  }
  if (threadIdx.x == 0) {
    float m   = ss[0] * (1.f / 65536.f);
    float var = qq[0] * (1.f / 65536.f) - m * m;
    meanr[bg] = m;
    rstdr[bg] = rsqrtf(var + 1e-5f);
  }
}

// ---------------------------------------------------------------------------
// GroupNorm pass 2 + swish, NCHW f32 -> NHWC f16 via LDS transpose tile.
// ---------------------------------------------------------------------------
__global__ __launch_bounds__(256)
void gn_apply(const float* __restrict__ x, const float* __restrict__ meanr,
              const float* __restrict__ rstdr, const float* __restrict__ gamma,
              const float* __restrict__ beta, _Float16* __restrict__ hs) {
  int xt = blockIdx.x & 3;
  int y  = (blockIdx.x >> 2) & 127;
  int b  = blockIdx.x >> 9;
  int x0 = xt * 32;

  __shared__ _Float16 tile[128][33];
  int t = threadIdx.x;
  int laneX = t & 31;
  int crow  = t >> 5;

  for (int it = 0; it < 16; ++it) {
    int c  = crow + it * 8;
    int bg = b * 32 + (c >> 2);
    float v  = x[(((size_t)(b * Cch) + c) * Hh + y) * Ww + x0 + laneX];
    float xn = (v - meanr[bg]) * rstdr[bg];
    float g  = xn * gamma[c] + beta[c];
    float sw = g / (1.f + expf(-g));
    tile[c][laneX] = (_Float16)sw;
  }
  __syncthreads();

  _Float16* dst = hs + (((size_t)(b * Hh) + y) * Ww + x0) * Cch;
  for (int it = 0; it < 16; ++it) {
    int idx = t + it * 256;
    int co  = idx & 127;
    int xi  = idx >> 7;
    dst[(size_t)xi * Cch + co] = tile[co][xi];
  }
}

// ---------------------------------------------------------------------------
// Pre-swizzle conv weights f32 OIHW -> f16 WMMA B-fragment layout.
// ---------------------------------------------------------------------------
__global__ __launch_bounds__(256)
void swz_weights(const float* __restrict__ w, _Float16* __restrict__ wf) {
  int id = blockIdx.x * 256 + threadIdx.x;
  if (id >= 147456) return;
  int e    = id & 15;
  int lane = (id >> 4) & 31;
  int coT  = (id >> 9) & 7;
  int kc   = (id >> 12) & 3;
  int tap  = id >> 14;
  int co = coT * 16 + (lane & 15);
  int k  = (e < 8) ? (lane >> 4) * 8 + e : 16 + (lane >> 4) * 8 + (e - 8);
  int ci = kc * 32 + k;
  int kh = tap / 3, kw = tap % 3;
  wf[id] = (_Float16)w[(((size_t)co * Cch + ci) * 3 + kh) * 3 + kw];
}

// ---------------------------------------------------------------------------
__global__ __launch_bounds__(128)
void nf_kernel(const float* __restrict__ temb, const float* __restrict__ nfw,
               const float* __restrict__ nfb, float* __restrict__ nf) {
  int b = blockIdx.x, c = threadIdx.x;
  float a = 0.f;
  for (int i = 0; i < 128; ++i) a += temb[b * 128 + i] * nfw[c * 128 + i];
  nf[b * 128 + c] = a + nfb[c];
}

__global__ __launch_bounds__(256)
void kern_kernel(const float* __restrict__ kv, const float* __restrict__ kW1,
                 const float* __restrict__ kW2, float* __restrict__ kern) {
  __shared__ float tbuf[64];
  int b = blockIdx.x;
  if (threadIdx.x < 64) {
    float a = 0.f;
    for (int i = 0; i < 64; ++i) a += kv[b * 64 + i] * kW1[threadIdx.x * 64 + i];
    tbuf[threadIdx.x] = (a >= 0.f) ? a : 0.1f * a;
  }
  __syncthreads();
  for (int o = threadIdx.x; o < 1152; o += 256) {
    float a = 0.f;
    for (int i = 0; i < 64; ++i) a += tbuf[i] * kW2[o * 64 + i];
    kern[(size_t)b * 1152 + o] = a;
  }
}

// ---------------------------------------------------------------------------
// TDM: 1-D row copy (32KB of f16) global -> LDS with per-pixel padding.
// D# per CDNA5 ISA:
//   g0: count=1 | lds_addr | global_addr(57b) | type=2
//   g1: data_size=1(2B), pad_enable, pad_interval=5 (64 dwords),
//       pad_amount=3 (4 dwords), tensor_dim0=tile_dim0=16384, tensor_dim1=1
// This toolchain exposes the 6-arg builtin (extra int32x8 group).
// ---------------------------------------------------------------------------
__device__ __forceinline__
void tdm_load_row(const _Float16* gsrc, unsigned lds_byte) {
  unsigned long long ga = (unsigned long long)(size_t)gsrc;
  uint32x4 g0;
  g0[0] = 1u;                                            // count=1
  g0[1] = lds_byte;                                      // lds_addr
  g0[2] = (unsigned)(ga & 0xFFFFFFFFu);                  // global_addr lo
  g0[3] = (unsigned)((ga >> 32) & 0x01FFFFFFu) | (2u << 30); // ga hi | type=2
  int32x8 g1;
  g1[0] = (int)((1u << 16) | (1u << 20) | (5u << 22) | (3u << 25));
  g1[1] = (int)((16384u & 0xFFFFu) << 16);               // tensor_dim0 lo16
  g1[2] = (int)((16384u >> 16) | (1u << 16));            // dim0 hi16 | dim1=1
  g1[3] = (int)(0x4000u << 16);                          // tile_dim0 = 16384
  g1[4] = 0;                                             // tile_dim1/2 unused
  g1[5] = 16384;                                         // dim0 stride
  g1[6] = 0;
  g1[7] = 0;
  int32x4 z4 = {0, 0, 0, 0};
  int32x8 z8 = {0, 0, 0, 0, 0, 0, 0, 0};
  __builtin_amdgcn_tensor_load_to_lds(g0, g1, z4, z4, z8, 0);
}

// ---------------------------------------------------------------------------
// Stage 4 input rows (y0-1 .. y0+2) into LDS; zero halo columns + OOB rows.
// ---------------------------------------------------------------------------
__device__ __forceinline__
void stage_rows(_Float16* smem, const _Float16* __restrict__ src, int b, int y0) {
  int tid = threadIdx.x;
  // zero halo columns xi=0 and xi=129 of each of the 4 rows
#pragma unroll
  for (int i = 0; i < 4; ++i) {
    int e = tid * 4 + i;                    // 0..1023
    int r = e >> 8, side = (e >> 7) & 1, c = e & 127;
    smem[(size_t)(r * ROW_PIX + side * 129) * PIX_STRIDE + c] = (_Float16)0.f;
  }
  if (y0 == 0) {
    for (int k = 0; k < 64; ++k) {
      int idx = tid + k * 256; int pix = idx >> 7, c = idx & 127;
      smem[(size_t)(1 + pix) * PIX_STRIDE + c] = (_Float16)0.f;
    }
  }
  if (y0 == Hh - 2) {
    for (int k = 0; k < 64; ++k) {
      int idx = tid + k * 256; int pix = idx >> 7, c = idx & 127;
      smem[(size_t)(3 * ROW_PIX + 1 + pix) * PIX_STRIDE + c] = (_Float16)0.f;
    }
  }
  if (tid < 32) {                            // wave 0 drives the TDM
    unsigned sbase = (unsigned)(size_t)smem; // LDS aperture: low 32 bits = offset
#pragma unroll
    for (int r = 0; r < 4; ++r) {
      int yy = y0 - 1 + r;
      if (yy >= 0 && yy < Hh) {
        const _Float16* gsrc = src + (((size_t)(b * Hh) + yy) * Ww) * Cch;
        tdm_load_row(gsrc, sbase + (unsigned)((r * ROW_PIX + 1) * PIX_STRIDE * 2));
      }
    }
    __builtin_amdgcn_s_wait_tensorcnt(0);
  }
  __syncthreads();
}

// ---------------------------------------------------------------------------
// Implicit-GEMM 3x3 conv core: one wave computes 32(pixels) x 128(co) using
// v_wmma_f32_16x16x32_f16.  A from LDS (branch-free, EXEC all-ones), B from
// global (L2-resident pre-swizzled fragments, each reused for 2 WMMAs).
// ---------------------------------------------------------------------------
__device__ __forceinline__
void conv_core(const _Float16* smem, const _Float16* __restrict__ wf,
               int rybase, int x0, int lane, v8f acc[2][8]) {
  int m = lane & 15, hf = lane >> 4;
#pragma unroll
  for (int tap = 0; tap < 9; ++tap) {
    int ry = rybase + tap / 3;
    int xbase = x0 + (tap % 3);             // xi for m=0 (halo offset +1, dx-1)
#pragma unroll
    for (int kc = 0; kc < 4; ++kc) {
      int ci = kc * 32 + hf * 8;
      v16h a[2];
#pragma unroll
      for (int f = 0; f < 2; ++f) {
        const _Float16* sp = smem + ((size_t)(ry * ROW_PIX + xbase + f * 16 + m) * PIX_STRIDE + ci);
        v8h lo = *(const v8h*)sp;
        v8h hi = *(const v8h*)(sp + 16);
#pragma unroll
        for (int i = 0; i < 8; ++i) { a[f][i] = lo[i]; a[f][i + 8] = hi[i]; }
      }
      const v16h* bptr = (const v16h*)wf + ((size_t)(tap * 4 + kc) * 8 * 32 + lane);
#pragma unroll
      for (int nt = 0; nt < 8; ++nt) {
        v16h bb = bptr[(size_t)nt * 32];
        acc[0][nt] = __builtin_amdgcn_wmma_f32_16x16x32_f16(
            false, a[0], false, bb, (short)0, acc[0][nt], false, false);
        acc[1][nt] = __builtin_amdgcn_wmma_f32_16x16x32_f16(
            false, a[1], false, bb, (short)0, acc[1][nt], false, false);
      }
    }
  }
}

// conv1: h = conv3x3(hs) + conv1_b + nf[b][c]   -> f16 NHWC
__global__ __launch_bounds__(256)
void conv1_wmma(const _Float16* __restrict__ hs, const _Float16* __restrict__ wf,
                const float* __restrict__ bias, const float* __restrict__ nf,
                _Float16* __restrict__ hout) {
  extern __shared__ _Float16 smem[];
  int b  = blockIdx.x >> 6;
  int y0 = (blockIdx.x & 63) * 2;
  __builtin_prefetch(wf + (size_t)threadIdx.x * 1152, 0, 3);
  stage_rows(smem, hs, b, y0);

  int wave = threadIdx.x >> 5, lane = threadIdx.x & 31;
  int yo = y0 + (wave >> 2);
  int x0 = (wave & 3) * 32;
  v8f acc[2][8] = {};
  conv_core(smem, wf, wave >> 2, x0, lane, acc);

  int n = lane & 15, hf = lane >> 4;
  _Float16* drow = hout + (((size_t)(b * Hh) + yo) * Ww) * Cch;
#pragma unroll
  for (int f = 0; f < 2; ++f)
#pragma unroll
    for (int nt = 0; nt < 8; ++nt) {
      int co = nt * 16 + n;
      float add = bias[co] + nf[b * 128 + co];
#pragma unroll
      for (int r = 0; r < 8; ++r) {
        int px = x0 + f * 16 + r + 8 * hf;
        drow[(size_t)px * Cch + co] = (_Float16)(acc[f][nt][r] + add);
      }
    }
}

// conv2: out(NCHW,f32) = conv3x3(h) + daconv_b + x   (depthwise added later)
__global__ __launch_bounds__(256)
void conv2_wmma(const _Float16* __restrict__ h, const _Float16* __restrict__ wf,
                const float* __restrict__ bias, const float* __restrict__ xin,
                float* __restrict__ out) {
  extern __shared__ _Float16 smem[];
  int b  = blockIdx.x >> 6;
  int y0 = (blockIdx.x & 63) * 2;
  __builtin_prefetch(wf + (size_t)threadIdx.x * 1152, 0, 3);
  stage_rows(smem, h, b, y0);

  int wave = threadIdx.x >> 5, lane = threadIdx.x & 31;
  int yo = y0 + (wave >> 2);
  int x0 = (wave & 3) * 32;
  v8f acc[2][8] = {};
  conv_core(smem, wf, wave >> 2, x0, lane, acc);

  int n = lane & 15, hf = lane >> 4;
#pragma unroll
  for (int f = 0; f < 2; ++f)
#pragma unroll
    for (int nt = 0; nt < 8; ++nt) {
      int co = nt * 16 + n;
      float bco = bias[co];
      const float* xrow = xin + (((size_t)(b * Cch) + co) * Hh + yo) * Ww;
      float*       orow = out + (((size_t)(b * Cch) + co) * Hh + yo) * Ww;
#pragma unroll
      for (int r = 0; r < 8; ++r) {
        int px = x0 + f * 16 + r + 8 * hf;
        orow[px] = acc[f][nt][r] + bco + xrow[px];
      }
    }
}

// Dynamic depthwise 3x3 + leaky-relu, accumulated into out (NCHW f32).
__global__ __launch_bounds__(256)
void dyn_conv(const _Float16* __restrict__ h, const float* __restrict__ kern,
              float* __restrict__ out) {
  int id = blockIdx.x * 256 + threadIdx.x;
  int c = id & 127;
  int x = (id >> 7) & 127;
  int y = (id >> 14) & 127;
  int b = id >> 21;
  const float* kn = kern + ((size_t)(b * Cch) + c) * 9;
  float a = 0.f;
#pragma unroll
  for (int tap = 0; tap < 9; ++tap) {
    int dy = tap / 3 - 1, dx = tap % 3 - 1;
    int yy = y + dy, xx = x + dx;
    if (yy >= 0 && yy < Hh && xx >= 0 && xx < Ww)
      a += kn[tap] * (float)h[(((size_t)(b * Hh) + yy) * Ww + xx) * Cch + c];
  }
  a = (a >= 0.f) ? a : 0.1f * a;
  out[(((size_t)(b * Cch) + c) * Hh + y) * Ww + x] += a;
}

// ---------------------------------------------------------------------------
extern "C" void kernel_launch(void* const* d_in, const int* in_sizes, int n_in,
                              void* d_out, int out_size, void* d_ws, size_t ws_size,
                              hipStream_t stream) {
  const float* x    = (const float*)d_in[0];
  const float* temb = (const float*)d_in[1];
  const float* kv   = (const float*)d_in[2];
  const float* g1   = (const float*)d_in[3];
  const float* b1   = (const float*)d_in[4];
  const float* w1   = (const float*)d_in[5];
  const float* cb1  = (const float*)d_in[6];
  const float* nfw  = (const float*)d_in[7];
  const float* nfb  = (const float*)d_in[8];
  const float* kW1  = (const float*)d_in[9];
  const float* kW2  = (const float*)d_in[10];
  const float* w2   = (const float*)d_in[11];
  const float* cb2  = (const float*)d_in[12];
  float* out = (float*)d_out;

  char* ws = (char*)d_ws;
  const size_t HS_BYTES = (size_t)Bsz * Hh * Ww * Cch * 2;   // 64 MB
  _Float16* hs   = (_Float16*)ws;
  _Float16* hbuf = (_Float16*)(ws + HS_BYTES);
  _Float16* wf1  = (_Float16*)(ws + 2 * HS_BYTES);
  _Float16* wf2  = (_Float16*)(ws + 2 * HS_BYTES + 294912);
  float* meanr = (float*)(ws + 2 * HS_BYTES + 2 * 294912 + 256);
  float* rstdr = meanr + 512;
  float* nf    = rstdr + 512;
  float* kern  = nf + 2048;

  const size_t SMEM_BYTES = (size_t)SMEM_HALVES * 2;          // 141440 B

  gn_stats <<<512, 256, 0, stream>>>(x, meanr, rstdr);
  gn_apply <<<Bsz * Hh * 4, 256, 0, stream>>>(x, meanr, rstdr, g1, b1, hs);
  swz_weights<<<576, 256, 0, stream>>>(w1, wf1);
  swz_weights<<<576, 256, 0, stream>>>(w2, wf2);
  nf_kernel <<<Bsz, 128, 0, stream>>>(temb, nfw, nfb, nf);
  kern_kernel<<<Bsz, 256, 0, stream>>>(kv, kW1, kW2, kern);
  conv1_wmma<<<Bsz * Hh / 2, 256, SMEM_BYTES, stream>>>(hs, wf1, cb1, nf, hbuf);
  conv2_wmma<<<Bsz * Hh / 2, 256, SMEM_BYTES, stream>>>(hbuf, wf2, cb2, x, out);
  dyn_conv  <<<(Bsz * Cch * Hh * Ww) / 256, 256, 0, stream>>>(hbuf, kern, out);
}
